// Decoder_17214228922493
// MI455X (gfx1250) — compile-verified, matrix operand order checked by reading, not run on hardware
//
#include <hip/hip_runtime.h>

typedef __attribute__((ext_vector_type(16))) __bf16 v16bf;
typedef __attribute__((ext_vector_type(8)))  float  v8f;
typedef unsigned int u32x4 __attribute__((ext_vector_type(4)));
typedef int          i32x4 __attribute__((ext_vector_type(4)));
typedef int          i32x8 __attribute__((ext_vector_type(8)));

#define EMB    512
#define INPUT  1024
#define HID    1536
#define SEQ    32
#define BATCH  64
#define GATES  (4 * HID)   // 6144

// Bytes of one (gate, jG) weight-fragment region: 48 kChunks * 32 lanes * 32B
#define WREG_BYTES (48 * 32 * 32)   // 49152

// ---------------------------------------------------------------------------
// Fragment layout (wave32, 16-bit WMMA, per ISA 7.12.2):
//   buffer of v16bf elements indexed [(group*kChunks + kChunk)*32 + lane]
//   lane L (0..15):  row = group*16 + L, halves hold K = {0..7, 16..23}+kChunk*32
//   lane L+16:       row = group*16 + L, halves hold K = {8..15, 24..31}+kChunk*32
// ---------------------------------------------------------------------------

__device__ __forceinline__ float sigmoidf(float x) {
    return 1.0f / (1.0f + expf(-x));
}

// TDM: DMA one contiguous 48KB weight region (as a 48 x 512-element 2D tile of
// 2-byte data) from global memory into LDS at lds_off. Tracked by TENSORcnt.
__device__ __forceinline__ void tdm_load_region(unsigned lds_off, const void* gptr) {
    unsigned long long ga = (unsigned long long)(uintptr_t)gptr;
    // D# group 0: count=1, lds_addr, 57-bit global_addr, type=2 ("image")
    u32x4 g0 = { 1u,
                 lds_off,
                 (unsigned)(ga & 0xFFFFFFFFu),
                 (unsigned)((ga >> 32) & 0x01FFFFFFu) | (2u << 30) };
    // D# group 1: workgroup_mask=0, data_size=1 (2B), no barrier/iterate/pad.
    //   tensor_dim0=512, tensor_dim1=48, tile_dim0=512, tile_dim1=48,
    //   tensor_dim0_stride=512.
    i32x8 g1 = { (1 << 16),        // data_size = 2 bytes
                 (512 << 16),      // tensor_dim0[15:0] in bits 63:48
                 (48  << 16),      // tensor_dim1[15:0] in bits 95:80
                 (512 << 16),      // tile_dim0 in bits 127:112
                 48,               // tile_dim1 in bits 143:128
                 512,              // tensor_dim0_stride[31:0]
                 0, 0 };
    i32x4 g2 = {};                 // dims 2/3 unused (tile_dim2 = 0)
    i32x4 g3 = {};
#if __has_include(<hip/amd_detail/amd_gfx1250_TDM.h>)
    i32x8 g4 = {};
    __builtin_amdgcn_tensor_load_to_lds(g0, g1, g2, g3, g4, 0);
#else
    __builtin_amdgcn_tensor_load_to_lds(g0, g1, g2, g3, 0);
#endif
}

// Pack row-major f32 [R x K] into bf16 WMMA fragment layout.
__global__ void pack_frags(const float* __restrict__ src, __bf16* __restrict__ dst,
                           int R, int K) {
    int kChunks = K >> 5;
    int tid = blockIdx.x * blockDim.x + threadIdx.x;
    int total = (R >> 4) * kChunks * 32;
    if (tid >= total) return;

    int lane   = tid & 31;
    int idx    = tid >> 5;               // (group*kChunks + kChunk)
    int kChunk = idx % kChunks;
    int group  = idx / kChunks;

    int r    = group * 16 + (lane & 15);
    int base = (lane >= 16) ? 8 : 0;

    v16bf frag;
#pragma unroll
    for (int h = 0; h < 16; ++h) {
        int k = kChunk * 32 + ((h < 8) ? (h + base) : (h + 8 + base));
        frag[h] = (__bf16)src[(size_t)r * K + k];
    }
    *(v16bf*)(dst + ((size_t)idx * 32 + lane) * 16) = frag;
}

// One-time: xg[64,6144] = x @ W_ih^T + b_ih + b_hh   (bf16 WMMA, f32 accum)
__global__ __launch_bounds__(128)
void xg_gemm(const __bf16* __restrict__ xFrag, const __bf16* __restrict__ wFrag,
             const float* __restrict__ b_ih, const float* __restrict__ b_hh,
             float* __restrict__ xg) {
    const int kChunks = INPUT / 32;                       // 32
    int wave = (blockIdx.x * blockDim.x + threadIdx.x) >> 5;
    int lane = threadIdx.x & 31;
    int mG = wave & 3;                                    // 0..3   (batch block)
    int nG = wave >> 2;                                   // 0..383 (gate-col block)

    const v16bf* aP = (const v16bf*)xFrag + (size_t)mG * kChunks * 32 + lane;
    const v16bf* bP = (const v16bf*)wFrag + (size_t)nG * kChunks * 32 + lane;

    v8f acc = {};
    for (int k = 0; k < kChunks; ++k) {
        if (k + 4 < kChunks) __builtin_prefetch((const void*)(bP + (k + 4) * 32), 0, 3);
        v16bf a = aP[k * 32];
        v16bf b = bP[k * 32];
        acc = __builtin_amdgcn_wmma_f32_16x16x32_bf16(false, a, false, b,
                                                      (short)0, acc, false, false);
    }

    int col = nG * 16 + (lane & 15);
    int hi  = (lane >> 4) ? 8 : 0;
    float bias = b_ih[col] + b_hh[col];
#pragma unroll
    for (int v = 0; v < 8; ++v) {
        int m = mG * 16 + v + hi;
        xg[(size_t)m * GATES + col] = acc[v] + bias;
    }
}

// One LSTM step. Block = 4 waves sharing one hidden-column block jG; the four
// gate weight regions (4 x 48KB) are TDM-DMA'd into LDS once per block, then
// each wave computes a 16(batch) x 16(hidden) tile of all four gates from LDS,
// applies the fused gating, updates c in place and writes h to preds[t].
__global__ __launch_bounds__(128)
void lstm_step(const __bf16* __restrict__ hFrag, const __bf16* __restrict__ wFrag,
               const float* __restrict__ xg, float* __restrict__ c,
               float* __restrict__ out) {
    extern __shared__ char smem[];                        // 4 * 49152 bytes
    const int kChunks = HID / 32;                         // 48
    int lane = threadIdx.x & 31;
    int mG = threadIdx.x >> 5;                            // 0..3  (batch block)
    int jG = blockIdx.x;                                  // 0..95 (hidden block)

    if (mG == 0) {
#pragma unroll
        for (int g = 0; g < 4; ++g) {
            const char* src = (const char*)wFrag + (size_t)(g * 96 + jG) * WREG_BYTES;
            tdm_load_region((unsigned)(g * WREG_BYTES), src);
        }
        __builtin_amdgcn_s_wait_tensorcnt(0);
    }
    __syncthreads();

    const v16bf* aP = (const v16bf*)hFrag + (size_t)mG * kChunks * 32 + lane;
    const v16bf* bL = (const v16bf*)smem;   // [g*48*32 + k*32 + lane]

    v8f ai = {}, af = {}, ag = {}, ao = {};
    for (int k = 0; k < kChunks; ++k) {
        if (k + 4 < kChunks) __builtin_prefetch((const void*)(aP + (k + 4) * 32), 0, 3);
        v16bf a = aP[k * 32];
        int kb = k * 32 + lane;
        ai = __builtin_amdgcn_wmma_f32_16x16x32_bf16(false, a, false, bL[0 * 1536 + kb],
                                                     (short)0, ai, false, false);
        af = __builtin_amdgcn_wmma_f32_16x16x32_bf16(false, a, false, bL[1 * 1536 + kb],
                                                     (short)0, af, false, false);
        ag = __builtin_amdgcn_wmma_f32_16x16x32_bf16(false, a, false, bL[2 * 1536 + kb],
                                                     (short)0, ag, false, false);
        ao = __builtin_amdgcn_wmma_f32_16x16x32_bf16(false, a, false, bL[3 * 1536 + kb],
                                                     (short)0, ao, false, false);
    }

    int col = lane & 15;
    int hi  = (lane >> 4) ? 8 : 0;
    int j   = jG * 16 + col;
#pragma unroll
    for (int v = 0; v < 8; ++v) {
        int m = mG * 16 + v + hi;
        size_t rowg = (size_t)m * GATES;
        float gi = ai[v] + xg[rowg + 0 * HID + j];
        float gf = af[v] + xg[rowg + 1 * HID + j];
        float gg = ag[v] + xg[rowg + 2 * HID + j];
        float go = ao[v] + xg[rowg + 3 * HID + j];

        size_t cidx = (size_t)m * HID + j;
        float cnew = sigmoidf(gf) * c[cidx] + sigmoidf(gi) * tanhf(gg);
        float hnew = sigmoidf(go) * tanhf(cnew);
        c[cidx]   = cnew;
        out[cidx] = hnew;     // out already offset to preds[t]
    }
}

extern "C" void kernel_launch(void* const* d_in, const int* in_sizes, int n_in,
                              void* d_out, int out_size, void* d_ws, size_t ws_size,
                              hipStream_t stream) {
    const float* seq = (const float*)d_in[0];   // [1,64,1024]
    const float* h0  = (const float*)d_in[1];   // [1,64,1536]
    const float* c0  = (const float*)d_in[2];   // [1,64,1536]
    const float* Wih = (const float*)d_in[3];   // [6144,1024]
    const float* Whh = (const float*)d_in[4];   // [6144,1536]
    const float* bih = (const float*)d_in[5];   // [6144]
    const float* bhh = (const float*)d_in[6];   // [6144]
    float* out = (float*)d_out;                 // [32,64,1536]

    char* ws = (char*)d_ws;
    size_t o = 0;
    __bf16* whhF = (__bf16*)(ws + o); o += (size_t)GATES * HID   * 2; // 18.9 MB
    __bf16* wihF = (__bf16*)(ws + o); o += (size_t)GATES * INPUT * 2; // 12.6 MB
    __bf16* xF   = (__bf16*)(ws + o); o += (size_t)BATCH * INPUT * 2; // 128 KB
    __bf16* hF   = (__bf16*)(ws + o); o += (size_t)BATCH * HID   * 2; // 192 KB
    float*  xgB  = (float*) (ws + o); o += (size_t)BATCH * GATES * 4; // 1.5 MB
    float*  cB   = (float*) (ws + o); o += (size_t)BATCH * HID   * 4; // 384 KB

    auto launchPack = [&](const float* src, __bf16* dst, int R, int K) {
        int total = (R / 16) * (K / 32) * 32;
        pack_frags<<<(total + 255) / 256, 256, 0, stream>>>(src, dst, R, K);
    };

    // One-time packing of weights and initial state into WMMA fragment layout.
    launchPack(Whh, whhF, GATES, HID);
    launchPack(Wih, wihF, GATES, INPUT);
    launchPack(seq, xF, BATCH, INPUT);
    launchPack(h0,  hF, BATCH, HID);
    hipMemcpyAsync(cB, c0, (size_t)BATCH * HID * sizeof(float),
                   hipMemcpyDeviceToDevice, stream);

    // xg = x @ W_ih^T + b_ih + b_hh : 1536 waves, 4 waves/block -> 384 blocks
    xg_gemm<<<384, 128, 0, stream>>>(xF, wihF, bih, bhh, xgB);

    // 32 sequential steps; stream ordering carries the recurrence.
    const unsigned smemBytes = 4 * WREG_BYTES;          // 192 KB of LDS per block
    for (int t = 0; t < SEQ; ++t) {
        float* outT = out + (size_t)t * BATCH * HID;
        lstm_step<<<96, 128, smemBytes, stream>>>(hF, whhF, xgB, cB, outT);
        if (t + 1 < SEQ)
            launchPack(outT, hF, BATCH, HID);   // repack h for next step's A
    }
}
